// QuantTFC_33801392619879
// MI455X (gfx1250) — 8-bit matrix operand order repaired and checked by reading, compiled, not run on hardware
//
#include <hip/hip_runtime.h>
#include <hip/hip_bf16.h>

typedef __attribute__((ext_vector_type(8))) int v8i;

#define BN_EPS 1e-5f

#define W0F_OFF 0u
#define W1F_OFF 196608u
#define W2F_OFF 200704u
#define W3F_OFF 204800u
#define PAR_OFF 205824u

__device__ __forceinline__ int frag_byte_off(int outch, int kk  ) {
  const int half = kk >> 5;
  const int l_hi = (kk >> 4) & 1;
  const int rem  = kk & 15;
  const int v    = half * 4 + (rem >> 2);
  const int b    = rem & 3;
  const int l    = l_hi * 16 + (outch & 15);
  return l * 32 + v * 4 + b;
}

__global__ void prep_pc(const float* __restrict__ W, const float* __restrict__ bias,
                        const float* __restrict__ g, const float* __restrict__ be,
                        const float* __restrict__ m, const float* __restrict__ v,
                        int F, int ntiles, float act_scale,
                        signed char* __restrict__ frag,
                        float* __restrict__ A, float* __restrict__ B) {
  __shared__ float red[256];
  const int row = blockIdx.x;
  const int t = threadIdx.x;
  const float* wr = W + (size_t)row * F;
  float mx = 0.f;
  for (int k = t; k < F; k += 256) mx = fmaxf(mx, fabsf(wr[k]));
  red[t] = mx; __syncthreads();
  for (int s = 128; s > 0; s >>= 1) { if (t < s) red[t] = fmaxf(red[t], red[t + s]); __syncthreads(); }
  const float s = fmaxf(red[0] * (1.f / 7.f), 1e-8f);
  const float inv_s = 1.f / s;
  const int nt = row >> 4;
  for (int k = t; k < F; k += 256) {
    int q = (int)rintf(wr[k] * inv_s);
    q = q > 7 ? 7 : (q < -7 ? -7 : q);
    const int ks = k >> 6, kk = k & 63;
    frag[(size_t)(ks * ntiles + nt) * 1024 + frag_byte_off(row, kk)] = (signed char)q;
  }
  if (t == 0) {
    const float rs = rsqrtf(v[row] + BN_EPS);
    const float alpha = g[row] * rs;
    A[row] = alpha * s * act_scale;
    B[row] = (bias[row] - m[row]) * alpha + be[row];
  }
}

__global__ void prep_pt(const float* __restrict__ W, const float* __restrict__ bias,
                        const float* __restrict__ g, const float* __restrict__ be,
                        const float* __restrict__ m, const float* __restrict__ v,
                        signed char* __restrict__ frag,
                        float* __restrict__ A, float* __restrict__ B) {
  __shared__ float red[256];
  const int t = threadIdx.x;
  float mx = 0.f;
  for (int k = t; k < 640; k += 256) mx = fmaxf(mx, fabsf(W[k]));
  red[t] = mx; __syncthreads();
  for (int s = 128; s > 0; s >>= 1) { if (t < s) red[t] = fmaxf(red[t], red[t + s]); __syncthreads(); }
  const float s = fmaxf(red[0] * (1.f / 7.f), 1e-8f);
  ((int*)frag)[t] = 0;
  __syncthreads();
  for (int idx = t; idx < 640; idx += 256) {
    const int row = idx >> 6, kk = idx & 63;
    int q = (int)rintf(W[idx] / s);
    q = q > 7 ? 7 : (q < -7 ? -7 : q);
    frag[frag_byte_off(row, kk)] = (signed char)q;
  }
  if (t < 16) {
    if (t < 10) {
      const float rs = rsqrtf(v[t] + BN_EPS);
      const float alpha = g[t] * rs;
      A[t] = alpha * s * (1.f / 7.f);
      B[t] = (bias[t] - m[t]) * alpha + be[t];
    } else { A[t] = 0.f; B[t] = 0.f; }
  }
}

__device__ __forceinline__ unsigned int quad_q8(float4 f) {
  float a = fminf(1.f, fmaxf(-1.f, f.x));
  float b = fminf(1.f, fmaxf(-1.f, f.y));
  float c = fminf(1.f, fmaxf(-1.f, f.z));
  float d = fminf(1.f, fmaxf(-1.f, f.w));
  const int q0 = (int)rintf(a * 127.f), q1 = (int)rintf(b * 127.f);
  const int q2 = (int)rintf(c * 127.f), q3 = (int)rintf(d * 127.f);
  return (unsigned)(q0 & 255) | ((unsigned)(q1 & 255) << 8) |
         ((unsigned)(q2 & 255) << 16) | ((unsigned)(q3 & 255) << 24);
}

__device__ __forceinline__ v8i load_frag(const signed char* base, int lane) {
  const uint4* p = (const uint4*)(base + lane * 32);
  const uint4 lo = p[0], hi = p[1];
  v8i r;
  r[0] = (int)lo.x; r[1] = (int)lo.y; r[2] = (int)lo.z; r[3] = (int)lo.w;
  r[4] = (int)hi.x; r[5] = (int)hi.y; r[6] = (int)hi.z; r[7] = (int)hi.w;
  return r;
}

__global__ __launch_bounds__(32) void tfc_fused(
    const float* __restrict__ x,
    const signed char* __restrict__ w0f, const signed char* __restrict__ w1f,
    const signed char* __restrict__ w2f, const signed char* __restrict__ w3f,
    const float* __restrict__ par, float* __restrict__ out) {
  __shared__ __align__(16) unsigned char act[1024];
  const int l = threadIdx.x;
  const int n = l & 15, lh = l >> 4;
  const int rowbase = blockIdx.x * 16;

  float a0[4], b0[4], a1[4], b1[4], a2[4], b2[4];
#pragma unroll
  for (int t = 0; t < 4; ++t) {
    a0[t] = par[  0 + t * 16 + n]; b0[t] = par[ 64 + t * 16 + n];
    a1[t] = par[128 + t * 16 + n]; b1[t] = par[192 + t * 16 + n];
    a2[t] = par[256 + t * 16 + n]; b2[t] = par[320 + t * 16 + n];
  }
  const float a3 = par[384 + n], b3 = par[400 + n];

  int aoff[8];
#pragma unroll
  for (int vv = 0; vv < 8; ++vv)
    aoff[vv] = (l & 15) * 64 + ((vv >> 2) * 32 + (vv & 3) * 4 + lh * 16);

  const int srow = l >> 1, shalf = l & 1;
  const float* xrow = x + (size_t)(rowbase + srow) * 3072 + shalf * 32;
  uint4* sdst = (uint4*)(act + srow * 64 + shalf * 32);

  v8i acc0 = {0,0,0,0,0,0,0,0}, acc1 = acc0, acc2 = acc0, acc3 = acc0;

  for (int ks = 0; ks < 48; ++ks) {
    if (ks + 1 < 48) __builtin_prefetch(xrow + (ks + 1) * 64, 0, 1);
    const float4* src = (const float4*)(xrow + ks * 64);
    unsigned int pk[8];
#pragma unroll
    for (int j = 0; j < 8; ++j) pk[j] = quad_q8(src[j]);
    sdst[0] = make_uint4(pk[0], pk[1], pk[2], pk[3]);
    sdst[1] = make_uint4(pk[4], pk[5], pk[6], pk[7]);
    __syncthreads();
    const signed char* wk = w0f + (size_t)ks * 4096;
    const v8i B0 = load_frag(wk,        l);
    const v8i B1 = load_frag(wk + 1024, l);
    const v8i B2 = load_frag(wk + 2048, l);
    const v8i B3 = load_frag(wk + 3072, l);
    v8i A;
#pragma unroll
    for (int vv = 0; vv < 8; ++vv) A[vv] = *(const int*)(act + aoff[vv]);
    acc0 = __builtin_amdgcn_wmma_i32_16x16x64_iu8(true, A, true, B0, acc0, false, false);
    acc1 = __builtin_amdgcn_wmma_i32_16x16x64_iu8(true, A, true, B1, acc1, false, false);
    acc2 = __builtin_amdgcn_wmma_i32_16x16x64_iu8(true, A, true, B2, acc2, false, false);
    acc3 = __builtin_amdgcn_wmma_i32_16x16x64_iu8(true, A, true, B3, acc3, false, false);
    __syncthreads();
  }

  auto emit_acts = [&](const v8i& c0, const v8i& c1, const v8i& c2, const v8i& c3,
                       const float* aa, const float* bb) {
#pragma unroll
    for (int r = 0; r < 8; ++r) {
      unsigned char* rowp = act + (r + 8 * lh) * 64 + n;
      rowp[ 0] = (unsigned char)(signed char)(int)rintf(fminf(1.f, fmaxf(-1.f, aa[0] * (float)c0[r] + bb[0])) * 7.f);
      rowp[16] = (unsigned char)(signed char)(int)rintf(fminf(1.f, fmaxf(-1.f, aa[1] * (float)c1[r] + bb[1])) * 7.f);
      rowp[32] = (unsigned char)(signed char)(int)rintf(fminf(1.f, fmaxf(-1.f, aa[2] * (float)c2[r] + bb[2])) * 7.f);
      rowp[48] = (unsigned char)(signed char)(int)rintf(fminf(1.f, fmaxf(-1.f, aa[3] * (float)c3[r] + bb[3])) * 7.f);
    }
  };

  const v8i zero = {0,0,0,0,0,0,0,0};

  emit_acts(acc0, acc1, acc2, acc3, a0, b0);
  __syncthreads();
  {
    const v8i B0 = load_frag(w1f,        l);
    const v8i B1 = load_frag(w1f + 1024, l);
    const v8i B2 = load_frag(w1f + 2048, l);
    const v8i B3 = load_frag(w1f + 3072, l);
    v8i A;
#pragma unroll
    for (int vv = 0; vv < 8; ++vv) A[vv] = *(const int*)(act + aoff[vv]);
    acc0 = __builtin_amdgcn_wmma_i32_16x16x64_iu8(true, A, true, B0, zero, false, false);
    acc1 = __builtin_amdgcn_wmma_i32_16x16x64_iu8(true, A, true, B1, zero, false, false);
    acc2 = __builtin_amdgcn_wmma_i32_16x16x64_iu8(true, A, true, B2, zero, false, false);
    acc3 = __builtin_amdgcn_wmma_i32_16x16x64_iu8(true, A, true, B3, zero, false, false);
  }
  __syncthreads();

  emit_acts(acc0, acc1, acc2, acc3, a1, b1);
  __syncthreads();
  {
    const v8i B0 = load_frag(w2f,        l);
    const v8i B1 = load_frag(w2f + 1024, l);
    const v8i B2 = load_frag(w2f + 2048, l);
    const v8i B3 = load_frag(w2f + 3072, l);
    v8i A;
#pragma unroll
    for (int vv = 0; vv < 8; ++vv) A[vv] = *(const int*)(act + aoff[vv]);
    acc0 = __builtin_amdgcn_wmma_i32_16x16x64_iu8(true, A, true, B0, zero, false, false);
    acc1 = __builtin_amdgcn_wmma_i32_16x16x64_iu8(true, A, true, B1, zero, false, false);
    acc2 = __builtin_amdgcn_wmma_i32_16x16x64_iu8(true, A, true, B2, zero, false, false);
    acc3 = __builtin_amdgcn_wmma_i32_16x16x64_iu8(true, A, true, B3, zero, false, false);
  }
  __syncthreads();

  emit_acts(acc0, acc1, acc2, acc3, a2, b2);
  __syncthreads();
  const v8i B3w = load_frag(w3f, l);
  v8i A;
#pragma unroll
  for (int vv = 0; vv < 8; ++vv) A[vv] = *(const int*)(act + aoff[vv]);
  const v8i o = __builtin_amdgcn_wmma_i32_16x16x64_iu8(true, A, true, B3w, zero, false, false);
  if (n < 10) {
#pragma unroll
    for (int r = 0; r < 8; ++r)
      out[(size_t)(rowbase + r + 8 * lh) * 10 + n] = a3 * (float)o[r] + b3;
  }
}

extern "C" void kernel_launch(void* const* d_in, const int* in_sizes, int n_in,
                              void* d_out, int out_size, void* d_ws, size_t ws_size,
                              hipStream_t stream) {
  (void)in_sizes; (void)n_in; (void)out_size; (void)ws_size;
  const float* x = (const float*)d_in[0];
  const float* W0 = (const float*)d_in[1];
  const float* p0[5] = {(const float*)d_in[2], (const float*)d_in[3], (const float*)d_in[4],
                        (const float*)d_in[5], (const float*)d_in[6]};
  const float* W1 = (const float*)d_in[7];
  const float* p1[5] = {(const float*)d_in[8], (const float*)d_in[9], (const float*)d_in[10],
                        (const float*)d_in[11], (const float*)d_in[12]};
  const float* W2 = (const float*)d_in[13];
  const float* p2[5] = {(const float*)d_in[14], (const float*)d_in[15], (const float*)d_in[16],
                        (const float*)d_in[17], (const float*)d_in[18]};
  const float* W3 = (const float*)d_in[19];
  const float* p3[5] = {(const float*)d_in[20], (const float*)d_in[21], (const float*)d_in[22],
                        (const float*)d_in[23], (const float*)d_in[24]};

  char* ws = (char*)d_ws;
  signed char* w0f = (signed char*)(ws + W0F_OFF);
  signed char* w1f = (signed char*)(ws + W1F_OFF);
  signed char* w2f = (signed char*)(ws + W2F_OFF);
  signed char* w3f = (signed char*)(ws + W3F_OFF);
  float* par = (float*)(ws + PAR_OFF);

  prep_pc<<<64, 256, 0, stream>>>(W0, p0[0], p0[1], p0[2], p0[3], p0[4],
                                  3072, 4, 1.f / 127.f, w0f, par + 0,   par + 64);
  prep_pc<<<64, 256, 0, stream>>>(W1, p1[0], p1[1], p1[2], p1[3], p1[4],
                                  64,   4, 1.f / 7.f,   w1f, par + 128, par + 192);
  prep_pc<<<64, 256, 0, stream>>>(W2, p2[0], p2[1], p2[2], p2[3], p2[4],
                                  64,   4, 1.f / 7.f,   w2f, par + 256, par + 320);
  prep_pt<<<1, 256, 0, stream>>>(W3, p3[0], p3[1], p3[2], p3[3], p3[4],
                                 w3f, par + 384, par + 400);

  tfc_fused<<<32768 / 16, 32, 0, stream>>>(x, w0f, w1f, w2f, w3f, par, (float*)d_out);
}
